// Block_52089363366667
// MI455X (gfx1250) — matrix operand order repaired and checked by reading, not compiled, not run on hardware
//
#include <hip/hip_runtime.h>
#include <stdint.h>
#include <stddef.h>

#define DIM      1024
#define D_INNER  2048
#define D_STATE  16
#define D_CONV   4
#define DT_RANK  64
#define BATCH    2
#define SEQLEN   2048
#define NTOK     (BATCH * SEQLEN)        // 4096 tokens
#define XDBL_W   (DT_RANK + 2 * D_STATE) // 96
#define EPS      1e-5f

typedef __bf16    bf16x16 __attribute__((ext_vector_type(16)));
typedef float     f32x8   __attribute__((ext_vector_type(8)));
typedef uint32_t  u32x4   __attribute__((ext_vector_type(4)));
typedef int       i32x4   __attribute__((ext_vector_type(4)));
typedef int       i32x8   __attribute__((ext_vector_type(8)));

// ---- toolchain detection for tensor_load_to_lds arity (ROCm7.2=5arg, therock=6arg)
#if defined(__has_include)
#if __has_include(<hip/amd_detail/amd_gfx1250_TDM.h>)
#define CDNA5_TDM_6ARG 1
#endif
#endif
#if !defined(CDNA5_TDM_6ARG) && (__clang_major__ >= 23)
#define CDNA5_TDM_6ARG 1
#endif

__device__ __forceinline__ uint16_t f32_to_bf16(float f) {
  uint32_t u = __float_as_uint(f);
  uint32_t r = (u + 0x7FFFu + ((u >> 16) & 1u)) >> 16;  // round-to-nearest-even
  return (uint16_t)r;
}

__device__ __forceinline__ void wait_tensorcnt0() {
#if defined(__has_builtin) && __has_builtin(__builtin_amdgcn_s_wait_tensorcnt)
  __builtin_amdgcn_s_wait_tensorcnt(0);
#else
  asm volatile("s_wait_tensorcnt 0x0" ::: "memory");
#endif
}

// TDM 2D tile load: rows x cols (bf16 elements) from row-major W (stride0 elems)
// into LDS at lds_off.  Pads +4 DWORDs after every 32 stored DWORDs (=one 64-el
// row), giving an LDS row stride of 72 elements (breaks bank conflicts).
// Descriptor per CDNA5 ISA 8.3/8.4 (count=1, data_size=2B, type=2).
__device__ __forceinline__ void tdm_load_2d(uint32_t lds_off,
                                            const uint16_t* gptr,
                                            uint32_t tensor_d0, uint32_t tensor_d1,
                                            uint32_t tile_d0,   uint32_t tile_d1,
                                            uint32_t stride0) {
  uint64_t ga = (uint64_t)(uintptr_t)gptr;
  u32x4 g0;
  g0[0] = 1u;                                                  // count=1
  g0[1] = lds_off;                                             // lds_addr
  g0[2] = (uint32_t)(ga & 0xffffffffu);                        // global_addr[31:0]
  g0[3] = (uint32_t)((ga >> 32) & 0x01ffffffu) | (2u << 30);   // ga[56:32] | type=2
  i32x8 g1;
  g1[0] = (int)((1u << 16) | (1u << 20) | (4u << 22) | (3u << 25));
          // wg_mask=0 | data_size=1(2B) | pad_enable | pad_interval=4(32dw) | pad_amount=3(4dw)
  g1[1] = (int)((tensor_d0 & 0xffffu) << 16);                  // tensor_dim0[15:0]
  g1[2] = (int)((tensor_d0 >> 16) | ((tensor_d1 & 0xffffu) << 16));
  g1[3] = (int)((tensor_d1 >> 16) | (tile_d0 << 16));          // | tile_dim0
  g1[4] = (int)(tile_d1 & 0xffffu);                            // tile_dim1, tile_dim2=0
  g1[5] = (int)stride0;                                        // tensor_dim0_stride
  g1[6] = 0;
  g1[7] = 0;
  i32x4 z4 = {0, 0, 0, 0};
#if defined(CDNA5_TDM_6ARG)
  i32x8 z8 = {0, 0, 0, 0, 0, 0, 0, 0};
  __builtin_amdgcn_tensor_load_to_lds(g0, g1, z4, z4, z8, 0);
#else
  __builtin_amdgcn_tensor_load_to_lds(g0, g1, z4, z4, 0);
#endif
}

// ---------------------------------------------------------------- fp32 -> bf16
__global__ void f2bf_kernel(const float* __restrict__ src,
                            uint16_t* __restrict__ dst, int n) {
  int i = blockIdx.x * blockDim.x + threadIdx.x;
  if (i < n) dst[i] = f32_to_bf16(src[i]);
}

// ------------------------------------------------- fused residual add + RMSNorm
__global__ void addnorm_kernel(const float* __restrict__ h,
                               const float* __restrict__ r,
                               const float* __restrict__ w,
                               const float* __restrict__ b,
                               float* __restrict__ res_out,
                               uint16_t* __restrict__ xb) {
  const int t   = blockIdx.x;
  const int tid = threadIdx.x;      // 256 threads
  const float* hrow = h + (size_t)t * DIM;
  const float* rrow = r + (size_t)t * DIM;
  float*       orow = res_out + (size_t)t * DIM;

  float v[DIM / 256];
  float ss = 0.f;
#pragma unroll
  for (int j = 0; j < DIM / 256; ++j) {
    int i = tid + j * 256;
    float x = hrow[i] + rrow[i];
    orow[i] = x;
    v[j] = x;
    ss += x * x;
  }
  __shared__ float sdata[256];
  sdata[tid] = ss;
  __syncthreads();
  for (int s = 128; s > 0; s >>= 1) {
    if (tid < s) sdata[tid] += sdata[tid + s];
    __syncthreads();
  }
  float rms = rsqrtf(sdata[0] / (float)DIM + EPS);
#pragma unroll
  for (int j = 0; j < DIM / 256; ++j) {
    int i = tid + j * 256;
    xb[(size_t)t * DIM + i] = f32_to_bf16(v[j] * rms * w[i] + b[i]);
  }
}

// ---------------------------------------------------------------- epilogue util
template <int EPI>
__device__ __forceinline__ void gemm_epilogue(f32x8* acc, int NTloc,
                                              int m0, int n0, int half, int lm,
                                              float* Cf, uint16_t* Cb,
                                              const float* bias, int ldc) {
  for (int t = 0; t < NTloc; ++t) {
#pragma unroll
    for (int r = 0; r < 8; ++r) {
      int row = m0 + r + half * 8;
      int col = n0 + t * 16 + lm;
      size_t idx = (size_t)row * ldc + col;
      float v = acc[t][r];
      if (EPI == 2) {
        float x = v + bias[col];
        Cf[idx] = (x > 20.f) ? x : __logf(1.f + __expf(x));   // softplus
      } else {
        Cf[idx] = v;
        if (EPI == 1) Cb[idx] = f32_to_bf16(v);
      }
    }
  }
}

// --------------------------- independent-wave bf16 WMMA GEMM (small-N: x_proj)
template <int NT, int EPI>
__global__ void gemm_bf16_kernel(const uint16_t* __restrict__ A, int lda,
                                 const uint16_t* __restrict__ W, int ldw,
                                 float* __restrict__ Cf,
                                 uint16_t* __restrict__ Cb,
                                 const float* __restrict__ bias,
                                 int M, int N, int K, int ldc) {
  const int lane    = threadIdx.x & 31;
  const int gwave   = (blockIdx.x * blockDim.x + threadIdx.x) >> 5;
  const int nGroups = N / (16 * NT);
  const int mTile   = gwave / nGroups;
  const int nGroup  = gwave % nGroups;
  if (mTile * 16 >= M) return;
  const int m0   = mTile * 16;
  const int n0   = nGroup * 16 * NT;
  const int half = lane >> 4;
  const int lm   = lane & 15;

  int koffA[8], koffB[8];
#pragma unroll
  for (int j = 0; j < 8; ++j) {
    koffA[j] = half * 16 + j * 2;
    koffB[j] = half * 16 + j * 2;
  }
  const uint16_t* arow = A + (size_t)(m0 + lm) * lda;
  const uint16_t* brow[NT];
#pragma unroll
  for (int t = 0; t < NT; ++t)
    brow[t] = W + (size_t)(n0 + t * 16 + lm) * ldw;

  f32x8 acc[NT];
#pragma unroll
  for (int t = 0; t < NT; ++t)
#pragma unroll
    for (int i = 0; i < 8; ++i) acc[t][i] = 0.f;

  for (int k = 0; k < K; k += 32) {
    union { bf16x16 v; uint32_t u[8]; } a;
#pragma unroll
    for (int j = 0; j < 8; ++j)
      a.u[j] = *(const uint32_t*)(arow + k + koffA[j]);
    union { bf16x16 v; uint32_t u[8]; } bm[NT];
#pragma unroll
    for (int t = 0; t < NT; ++t)
#pragma unroll
      for (int j = 0; j < 8; ++j)
        bm[t].u[j] = *(const uint32_t*)(brow[t] + k + koffB[j]);
#pragma unroll
    for (int t = 0; t < NT; ++t)
      acc[t] = __builtin_amdgcn_wmma_f32_16x16x32_bf16(
          false, a.v, false, bm[t].v, (short)0, acc[t], false, false);
  }
  gemm_epilogue<EPI>(acc, NT, m0, n0, half, lm, Cf, Cb, bias, ldc);
}

// --------------- block-cooperative WMMA GEMM, B staged in LDS by the TDM
// Block = 4 waves; each wave owns a 16-row M tile; all share a 64-wide N group.
// B tile (64 rows x 64 K, bf16) double-buffered in LDS via tensor_load_to_lds.
template <int EPI>
__global__ void gemm_bf16_tdm_kernel(const uint16_t* __restrict__ A, int lda,
                                     const uint16_t* __restrict__ W, int ldw,
                                     float* __restrict__ Cf,
                                     uint16_t* __restrict__ Cb,
                                     const float* __restrict__ bias,
                                     int M, int N, int K, int ldc) {
  constexpr int NT   = 4;    // 64-wide N tile
  constexpr int KC   = 64;   // K chunk
  constexpr int ROWP = 72;   // padded LDS row stride (TDM pads 4dw per 32dw row)
  __shared__ uint16_t btile[2][64 * ROWP];

  const int lane   = threadIdx.x & 31;
  const int wave   = threadIdx.x >> 5;
  const int nGroups = N / 64;
  const int mBlock  = blockIdx.x / nGroups;
  const int nGroup  = blockIdx.x % nGroups;
  const int m0   = mBlock * 64 + wave * 16;
  const int n0   = nGroup * 64;
  const int half = lane >> 4;
  const int lm   = lane & 15;

  int koffA[8];
#pragma unroll
  for (int j = 0; j < 8; ++j)
    koffA[j] = half * 16 + j * 2;
  const uint16_t* arow = A + (size_t)(m0 + lm) * lda;

  f32x8 acc[NT];
#pragma unroll
  for (int t = 0; t < NT; ++t)
#pragma unroll
    for (int i = 0; i < 8; ++i) acc[t][i] = 0.f;

  const int nChunks = K / KC;
  if (wave == 0) {   // prologue: stage chunk 0
    tdm_load_2d((uint32_t)(uintptr_t)&btile[0][0],
                W + (size_t)n0 * ldw, (uint32_t)ldw, (uint32_t)N,
                KC, 64, (uint32_t)ldw);
    wait_tensorcnt0();
  }
  __syncthreads();

  for (int c = 0; c < nChunks; ++c) {
    const int kbase = c * KC;
    const int cur   = c & 1;
    if (wave == 0 && c + 1 < nChunks) {   // async-stage next chunk
      tdm_load_2d((uint32_t)(uintptr_t)&btile[cur ^ 1][0],
                  W + (size_t)n0 * ldw + (kbase + KC), (uint32_t)ldw, (uint32_t)N,
                  KC, 64, (uint32_t)ldw);
    }
    if (c + 1 < nChunks)
      __builtin_prefetch(arow + kbase + KC, 0, 1);   // next A chunk -> caches

    const uint16_t* bt = btile[cur];
#pragma unroll
    for (int kk = 0; kk < KC; kk += 32) {
      union { bf16x16 v; uint32_t u[8]; } a;
#pragma unroll
      for (int j = 0; j < 8; ++j)
        a.u[j] = *(const uint32_t*)(arow + kbase + kk + koffA[j]);
      // batch all NT B-fragment LDS reads, then issue the 4 WMMAs back-to-back
      union { bf16x16 v; u32x4 q[2]; } bm[NT];
#pragma unroll
      for (int t = 0; t < NT; ++t) {
        const uint16_t* lrow = bt + (t * 16 + lm) * ROWP + kk + half * 16;
        bm[t].q[0] = *(const u32x4*)(lrow);
        bm[t].q[1] = *(const u32x4*)(lrow + 8);
      }
#pragma unroll
      for (int t = 0; t < NT; ++t)
        acc[t] = __builtin_amdgcn_wmma_f32_16x16x32_bf16(
            false, a.v, false, bm[t].v, (short)0, acc[t], false, false);
    }
    if (wave == 0 && c + 1 < nChunks) wait_tensorcnt0();
    __syncthreads();
  }
  gemm_epilogue<EPI>(acc, NT, m0, n0, half, lm, Cf, Cb, bias, ldc);
}

// --------------------------------------- depthwise causal conv(4) + bias + SiLU
__global__ void conv_silu_kernel(const float* __restrict__ xz,
                                 const float* __restrict__ cw,
                                 const float* __restrict__ cb,
                                 float* __restrict__ xif,
                                 uint16_t* __restrict__ xib) {
  int idx = blockIdx.x * blockDim.x + threadIdx.x;
  if (idx >= NTOK * D_INNER) return;
  int d  = idx % D_INNER;
  int tl = idx / D_INNER;
  int l  = tl % SEQLEN;
  float acc = cb[d];
#pragma unroll
  for (int j = 0; j < D_CONV; ++j) {
    int ls = l + j - (D_CONV - 1);
    if (ls >= 0)
      acc += cw[d * D_CONV + j] *
             xz[(size_t)(tl + j - (D_CONV - 1)) * (2 * D_INNER) + d];
  }
  float s = acc / (1.f + __expf(-acc));   // SiLU
  xif[idx] = s;
  xib[idx] = f32_to_bf16(s);
}

// --- selective scan: 16 lanes per (b,d) channel, shfl_xor butterfly over state
__global__ void scan_kernel(const float* __restrict__ dt,
                            const float* __restrict__ xdbl,
                            const float* __restrict__ xif,
                            const float* __restrict__ xz,
                            const float* __restrict__ A_log,
                            const float* __restrict__ Dp,
                            uint16_t* __restrict__ yb) {
  int gt = blockIdx.x * blockDim.x + threadIdx.x;
  int ch = gt >> 4;                  // b * D_INNER + d
  int n  = gt & 15;                  // state index
  if (ch >= BATCH * D_INNER) return;
  int d = ch % D_INNER;
  int b = ch / D_INNER;

  const float An = -__expf(A_log[d * D_STATE + n]);
  const float Dd = Dp[d];
  float h = 0.f;

  for (int l = 0; l < SEQLEN; ++l) {
    size_t tl = (size_t)b * SEQLEN + l;
    float dtv = dt[tl * D_INNER + d];          // uniform across 16-lane group
    float xiv = xif[tl * D_INNER + d];
    const float* row = xdbl + tl * XDBL_W;
    float dA = __expf(dtv * An);
    h = dA * h + (dtv * xiv) * row[DT_RANK + n];          // B_t[n]
    float p = h * row[DT_RANK + D_STATE + n];             // C_t[n]
    p += __shfl_xor(p, 1);                                // reduce 16 states
    p += __shfl_xor(p, 2);
    p += __shfl_xor(p, 4);
    p += __shfl_xor(p, 8);
    if (n == 0) {
      float y = p + Dd * xiv;
      float z = xz[tl * (2 * D_INNER) + D_INNER + d];
      y *= z / (1.f + __expf(-z));                        // * SiLU(z)
      yb[tl * D_INNER + d] = f32_to_bf16(y);
    }
  }
}

// ------------------------------------------------------------------------------
extern "C" void kernel_launch(void* const* d_in, const int* in_sizes, int n_in,
                              void* d_out, int out_size, void* d_ws, size_t ws_size,
                              hipStream_t stream) {
  const float* hid   = (const float*)d_in[0];
  const float* res   = (const float*)d_in[1];
  const float* nw    = (const float*)d_in[2];
  const float* nb    = (const float*)d_in[3];
  const float* w_in  = (const float*)d_in[4];
  const float* cw    = (const float*)d_in[5];
  const float* cbp   = (const float*)d_in[6];
  const float* w_x   = (const float*)d_in[7];
  const float* w_dt  = (const float*)d_in[8];
  const float* b_dt  = (const float*)d_in[9];
  const float* alog  = (const float*)d_in[10];
  const float* dpar  = (const float*)d_in[11];
  const float* w_out = (const float*)d_in[12];

  float* out     = (float*)d_out;                  // [NTOK, DIM]
  float* res_out = out + (size_t)NTOK * DIM;       // [NTOK, DIM]

  char* ws = (char*)d_ws;
  size_t off = 0;
  auto alloc = [&](size_t bytes) -> void* {
    void* p = ws + off;
    off += (bytes + 255) & ~(size_t)255;
    return p;
  };
  uint16_t* x_bf    = (uint16_t*)alloc((size_t)NTOK * DIM * 2);
  uint16_t* w_in_b  = (uint16_t*)alloc((size_t)2 * D_INNER * DIM * 2);
  uint16_t* w_x_b   = (uint16_t*)alloc((size_t)XDBL_W * D_INNER * 2);
  uint16_t* w_dt_b  = (uint16_t*)alloc((size_t)D_INNER * DT_RANK * 2);
  uint16_t* w_out_b = (uint16_t*)alloc((size_t)DIM * D_INNER * 2);
  float*    xz      = (float*)   alloc((size_t)NTOK * 2 * D_INNER * 4);
  float*    xi_f    = (float*)   alloc((size_t)NTOK * D_INNER * 4);
  uint16_t* xi_b    = (uint16_t*)alloc((size_t)NTOK * D_INNER * 2);
  float*    xdbl_f  = (float*)   alloc((size_t)NTOK * XDBL_W * 4);
  uint16_t* xdbl_b  = (uint16_t*)alloc((size_t)NTOK * XDBL_W * 2);
  float*    dt_f    = (float*)   alloc((size_t)NTOK * D_INNER * 4);
  uint16_t* y_b     = (uint16_t*)alloc((size_t)NTOK * D_INNER * 2);
  (void)ws_size; (void)n_in; (void)in_sizes; (void)out_size;

  // 0) weights -> bf16
  int n1 = 2 * D_INNER * DIM;
  int n2 = XDBL_W * D_INNER;
  int n3 = D_INNER * DT_RANK;
  int n4 = DIM * D_INNER;
  f2bf_kernel<<<(n1 + 255) / 256, 256, 0, stream>>>(w_in,  w_in_b,  n1);
  f2bf_kernel<<<(n2 + 255) / 256, 256, 0, stream>>>(w_x,   w_x_b,   n2);
  f2bf_kernel<<<(n3 + 255) / 256, 256, 0, stream>>>(w_dt,  w_dt_b,  n3);
  f2bf_kernel<<<(n4 + 255) / 256, 256, 0, stream>>>(w_out, w_out_b, n4);

  // 1) residual add + RMSNorm (also writes second tuple output)
  addnorm_kernel<<<NTOK, 256, 0, stream>>>(hid, res, nw, nb, res_out, x_bf);

  // 2) in_proj (TDM-staged): [4096,1024] x [4096,1024]^T -> xz [4096,4096]
  {
    int M = NTOK, N = 2 * D_INNER, K = DIM;
    gemm_bf16_tdm_kernel<0><<<(M / 64) * (N / 64), 128, 0, stream>>>(
        x_bf, DIM, w_in_b, DIM, xz, nullptr, nullptr, M, N, K, N);
  }

  // 3) depthwise causal conv + SiLU
  conv_silu_kernel<<<(NTOK * D_INNER + 255) / 256, 256, 0, stream>>>(
      xz, cw, cbp, xi_f, xi_b);

  // 4) x_proj (N=96): -> x_dbl [4096,96] (fp32 + bf16 mirror)
  {
    int M = NTOK, N = XDBL_W, K = D_INNER;
    int waves = (M / 16) * (N / 32);
    gemm_bf16_kernel<2, 1><<<waves / 4, 128, 0, stream>>>(
        xi_b, D_INNER, w_x_b, D_INNER, xdbl_f, xdbl_b, nullptr, M, N, K, N);
  }

  // 5) dt_proj (+bias, softplus; TDM-staged): A = x_dbl[:, :64] (lda=96), K=64
  {
    int M = NTOK, N = D_INNER, K = DT_RANK;
    gemm_bf16_tdm_kernel<2><<<(M / 64) * (N / 64), 128, 0, stream>>>(
        xdbl_b, XDBL_W, w_dt_b, DT_RANK, dt_f, nullptr, b_dt, M, N, K, N);
  }

  // 6) selective scan + gating (16 lanes per channel)
  scan_kernel<<<(BATCH * D_INNER * 16) / 256, 256, 0, stream>>>(
      dt_f, xdbl_f, xi_f, xz, alog, dpar, y_b);

  // 7) out_proj (TDM-staged) -> first tuple output
  {
    int M = NTOK, N = DIM, K = D_INNER;
    gemm_bf16_tdm_kernel<0><<<(M / 64) * (N / 64), 128, 0, stream>>>(
        y_b, D_INNER, w_out_b, D_INNER, out, nullptr, nullptr, M, N, K, N);
  }
}
